// SpikeFPN_GAD_84078279786805
// MI455X (gfx1250) — compile-verified, matrix operand order checked by reading, not run on hardware
//
#include <hip/hip_runtime.h>
#include <hip/hip_bf16.h>
#include <stdint.h>

typedef __attribute__((ext_vector_type(16))) _Float16 v16h;
typedef __attribute__((ext_vector_type(8)))  float    v8f;

#define CIN   5
#define COUT  5
#define KTOT  45          // CIN*3*3
#define IMG   512
#define TY    16          // tile rows per workgroup
#define TX    64          // tile cols per workgroup
#define IH    18          // TY + 2 halo
#define IWP   68          // TX + 2 halo, padded
#define CINP  8           // channel slots padded to 8 so K=45..63 reads hit zeroed LDS
#define ENC_ELEMS 52428800ULL   // 8*5*5*512*512

// ---------------------------------------------------------------- helpers ---

__device__ __forceinline__ uint32_t lds_off_u32(const void* p) {
  // low 32 bits of a flat LDS address == wave-relative LDS byte offset
  return (uint32_t)(uintptr_t)p;
}

__device__ __forceinline__ void async_ld_b32(uint32_t ldsoff, const float* g) {
  // CDNA5: GLOBAL_LOAD_ASYNC_TO_LDS_B32  (VDST = LDS byte offset, VADDR = 64b global)
  asm volatile("global_load_async_to_lds_b32 %0, %1, off"
               :: "v"(ldsoff), "v"(g) : "memory");
}

__device__ __forceinline__ void wait_async0() {
  asm volatile("s_wait_asynccnt 0" ::: "memory");
}

// rcp-based sigmoid: v_exp_f32 + v_add + v_rcp_f32 (avoids IEEE div expansion)
__device__ __forceinline__ float fast_sigmoid(float v) {
  return __builtin_amdgcn_rcpf(1.0f + __expf(-v));
}

// map flattened patch index k = c*9 + ky*3 + kx -> LDS float offset inside tile
__device__ __forceinline__ int patch_off(int k) {
  int c  = k / 9;
  int r  = k - c * 9;
  int ky = r / 3;
  int kx = r - ky * 3;
  return (c * IH + ky) * IWP + kx;
}

// ------------------------------------------------- kernel 1: conv + LTC -----
// Implicit-GEMM conv: D[cout(16), x(16)] = W'[cout, k(64)] x Patch[k, x]
// A = weights (bn_gamma folded), built once; B gathered from async-staged LDS tile.

__global__ __launch_bounds__(256) void conv_ltc_kernel(
    const float* __restrict__ x,     const float* __restrict__ convw,
    const float* __restrict__ gamma, const float* __restrict__ beta,
    const float* __restrict__ cm,    const float* __restrict__ vleak,
    const float* __restrict__ tau_m, const float* __restrict__ erev,
    float* __restrict__ enc)
{
  __shared__ float sIn[CINP * IH * IWP];   // 39,168 B (padded channels pre-zeroed)
  __shared__ float sVp[COUT * TY * TX];    // 20,480 B (LTC state across T)

  const int tid  = threadIdx.x;
  const int lane = tid & 31;
  const int wave = tid >> 5;
  const int xt = blockIdx.x, yt = blockIdx.y, b = blockIdx.z;
  const int x0 = xt * TX, y0 = yt * TY;

  // zero LDS: covers conv zero-padding halo + padded K channels + v0 = 0
  for (int i = tid; i < CINP * IH * IWP; i += 256) sIn[i] = 0.f;
  for (int i = tid; i < COUT * TY * TX; i += 256) sVp[i] = 0.f;

  // uniform per-channel LTC params
  float Pb[COUT], Pc[COUT], Pv[COUT], Pt[COUT], Pe[COUT];
#pragma unroll
  for (int g = 0; g < COUT; ++g) {
    Pb[g] = beta[g]; Pc[g] = cm[g]; Pv[g] = vleak[g]; Pt[g] = tau_m[g]; Pe[g] = erev[g];
  }

  const int m  = lane & 15;   // A row (cout) / B col (pixel)
  const int kh = lane >> 4;   // K-half select per ISA f16 layout
  const int n  = lane & 15;

  // A fragments (16x32 f16, two K-fragments covering K=0..63), gamma folded in
  const float gm = (m < COUT) ? gamma[m] : 0.f;
  v16h afrag[2];
#pragma unroll
  for (int f = 0; f < 2; ++f)
#pragma unroll
    for (int j = 0; j < 8; ++j)
#pragma unroll
      for (int e = 0; e < 2; ++e) {
        int k = f * 32 + ((j >> 2) << 4) + (kh << 3) + ((j & 3) << 1) + e;
        float wv = (m < COUT && k < KTOT) ? convw[m * KTOT + k] * gm : 0.f;
        afrag[f][2 * j + e] = (_Float16)wv;
      }

  // B gather offsets (float-index, two packed 16-bit offsets per VGPR pair)
  uint32_t offp[2][8];
#pragma unroll
  for (int f = 0; f < 2; ++f)
#pragma unroll
    for (int j = 0; j < 8; ++j) {
      int k0 = f * 32 + ((j >> 2) << 4) + (kh << 3) + ((j & 3) << 1);
      offp[f][j] = (uint32_t)patch_off(k0) | ((uint32_t)patch_off(k0 + 1) << 16);
    }

  __syncthreads();

  for (int t = 0; t < 5; ++t) {
    const float* xt_ = x + (((size_t)b * 5 + t) * CIN) * (size_t)(IMG * IMG);

    // async-stage input tile (with halo) into LDS; OOB stays zero.
    // Also prefetch next timestep's tile into cache while this one is consumed.
    for (int idx = tid; idx < CIN * IH * 66; idx += 256) {
      int rx = idx % 66;
      int t2 = idx / 66;
      int ry = t2 % IH;
      int c  = t2 / IH;
      int gy = y0 - 1 + ry, gx = x0 - 1 + rx;
      if ((unsigned)gy < IMG && (unsigned)gx < IMG) {
        const float* gp = xt_ + (size_t)c * IMG * IMG + (size_t)gy * IMG + gx;
        uint32_t lo = lds_off_u32(&sIn[(c * IH + ry) * IWP + rx]);
        async_ld_b32(lo, gp);
        if (t < 4) __builtin_prefetch(gp + (size_t)CIN * IMG * IMG, 0, 0);
      }
    }
    wait_async0();
    __syncthreads();

    // 64 chunks of 16 output pixels; 8 per wave
    for (int i = 0; i < 8; ++i) {
      int chunk = wave * 8 + i;
      int yl  = chunk >> 2;
      int xb4 = (chunk & 3) << 4;
      int basef = yl * IWP + xb4 + n;

      v8f acc = {};
#pragma unroll
      for (int f = 0; f < 2; ++f) {
        v16h bfrag;
#pragma unroll
        for (int j = 0; j < 8; ++j) {
          uint32_t op = offp[f][j];
          float f0 = sIn[basef + (op & 0xffffu)];
          float f1 = sIn[basef + (op >> 16)];
          bfrag[2 * j]     = (_Float16)f0;
          bfrag[2 * j + 1] = (_Float16)f1;
        }
        acc = __builtin_amdgcn_wmma_f32_16x16x32_f16(
            false, afrag[f], false, bfrag, (short)0, acc, false, false);
      }

      // LTC recurrence (cout 0..4 live in acc VGPRs 0..4, lanes 0..15).
      // Note: fp32 (1.0 + 1e-8) == 1.0, so the reference's /(1+EPS) is identity.
      const int vbase = yl * TX + xb4 + n;
      float vals[COUT];
#pragma unroll
      for (int g = 0; g < COUT; ++g) {
        float wih = acc[g] + Pb[g];
        float sg  = fast_sigmoid(wih);
        float vp  = sVp[g * (TY * TX) + vbase];
        float num = Pt[g] * vp * __builtin_amdgcn_rcpf(Pv[g] + Pc[g] * sg)
                    + wih * Pe[g];
        vals[g] = fast_sigmoid(num);
      }
      if (lane < 16) {   // single EXEC-guarded store block per chunk
        size_t ebase = ((((size_t)b * 5 + t) * COUT) * IMG + (y0 + yl)) * IMG
                       + (x0 + xb4 + n);
#pragma unroll
        for (int g = 0; g < COUT; ++g) {
          sVp[g * (TY * TX) + vbase] = vals[g];
          enc[ebase + (size_t)g * IMG * IMG] = vals[g];
        }
      }
    }
    __syncthreads();   // all reads done before next tile overwrite
  }
}

// ------------------------------------------------- kernel 2: 1x1-conv head --
// out[o, p] = sum_c w[o,c] * y[b,c,p] + bias[o]; M=21 padded to 32 (2 acc tiles)

__global__ __launch_bounds__(256) void head_kernel(
    const float* __restrict__ y, const float* __restrict__ w,
    const float* __restrict__ bias, int C, int HW, float* __restrict__ pred)
{
  const int tid = threadIdx.x, lane = tid & 31, wave = tid >> 5;
  const int b  = blockIdx.z;
  const int p0 = blockIdx.x * 128 + wave * 16;
  const int n  = lane & 15;
  const int kh = lane >> 4;
  const int m  = lane & 15;
  const float* yb = y + (size_t)b * C * HW;

  v8f acc0 = {}, acc1 = {};
  for (int kb = 0; kb < C; kb += 32) {
    v16h af0, af1, bf;
#pragma unroll
    for (int j = 0; j < 8; ++j) {
      int k0 = kb + ((j >> 2) << 4) + (kh << 3) + ((j & 3) << 1);
      bf[2 * j]     = (_Float16)yb[(size_t)k0 * HW + p0 + n];
      bf[2 * j + 1] = (_Float16)yb[(size_t)(k0 + 1) * HW + p0 + n];
      af0[2 * j]     = (_Float16)w[m * C + k0];
      af0[2 * j + 1] = (_Float16)w[m * C + k0 + 1];
      int m1 = 16 + m;
      float w0 = (m1 < 21) ? w[m1 * C + k0]     : 0.f;
      float w1 = (m1 < 21) ? w[m1 * C + k0 + 1] : 0.f;
      af1[2 * j]     = (_Float16)w0;
      af1[2 * j + 1] = (_Float16)w1;
    }
    acc0 = __builtin_amdgcn_wmma_f32_16x16x32_f16(false, af0, false, bf, (short)0, acc0, false, false);
    acc1 = __builtin_amdgcn_wmma_f32_16x16x32_f16(false, af1, false, bf, (short)0, acc1, false, false);
  }

  float* pb = pred + (size_t)b * 21 * HW;
#pragma unroll
  for (int g = 0; g < 8; ++g) {
    int mr = g + ((lane >> 4) << 3);   // C/D layout: M = g (+8 for upper lanes)
    int px = p0 + n;
    pb[mr * HW + px] = acc0[g] + bias[mr];               // rows 0..15
    int o1 = 16 + mr;
    if (o1 < 21) pb[o1 * HW + px] = acc1[g] + bias[o1];  // rows 16..20
  }
}

// ------------------------------------------------- kernel 3: decode ---------

__device__ const float g_anchors[3][3][2] = {
  {{10.f, 13.f}, {16.f, 30.f}, {33.f, 23.f}},
  {{30.f, 61.f}, {62.f, 45.f}, {59.f, 119.f}},
  {{116.f, 90.f}, {156.f, 198.f}, {373.f, 326.f}}};

__global__ __launch_bounds__(256) void decode_kernel(
    const float* __restrict__ ws, float* __restrict__ det)
{
  int gid = blockIdx.x * 256 + threadIdx.x;
  if (gid >= 8 * 16128) return;
  int b = gid / 16128;
  int r = gid - b * 16128;

  int scale, base, fdim, stride;
  const float* pred;
  if (r < 12288)      { scale = 0; base = 0;     fdim = 64; stride = 8;  pred = ws; }
  else if (r < 15360) { scale = 1; base = 12288; fdim = 32; stride = 16; pred = ws + 688128; }
  else                { scale = 2; base = 15360; fdim = 16; stride = 32; pred = ws + 860160; }

  int local = r - base;
  int pix = local / 3, a = local - pix * 3;
  int gy = pix / fdim, gx = pix - gy * fdim;
  int HW = fdim * fdim;
  const float* pb = pred + (size_t)b * 21 * HW;

  float obj = pb[a * HW + pix];
  float c0  = pb[(3 + a * 2 + 0) * HW + pix];
  float c1  = pb[(3 + a * 2 + 1) * HW + pix];
  float rx  = pb[(9 + a * 4 + 0) * HW + pix];
  float ry  = pb[(9 + a * 4 + 1) * HW + pix];
  float rw  = pb[(9 + a * 4 + 2) * HW + pix];
  float rh  = pb[(9 + a * 4 + 3) * HW + pix];

  float cx = (fast_sigmoid(rx) + (float)gx) * (float)stride;
  float cy = (fast_sigmoid(ry) + (float)gy) * (float)stride;
  float wwv = __expf(rw) * g_anchors[scale][a][0];
  float hhv = __expf(rh) * g_anchors[scale][a][1];

  float* o = det + ((size_t)b * 16128 + r) * 7;
  o[0] = obj; o[1] = c0; o[2] = c1;
  o[3] = (cx - wwv * 0.5f) * (1.0f / 512.f);
  o[4] = (cy - hhv * 0.5f) * (1.0f / 512.f);
  o[5] = (cx + wwv * 0.5f) * (1.0f / 512.f);
  o[6] = (cy + hhv * 0.5f) * (1.0f / 512.f);
}

// ------------------------------------------------- launch -------------------

extern "C" void kernel_launch(void* const* d_in, const int* in_sizes, int n_in,
                              void* d_out, int out_size, void* d_ws, size_t ws_size,
                              hipStream_t stream) {
  (void)in_sizes; (void)n_in; (void)out_size; (void)ws_size;
  const float* x     = (const float*)d_in[0];
  const float* y1    = (const float*)d_in[1];
  const float* y2    = (const float*)d_in[2];
  const float* y3    = (const float*)d_in[3];
  const float* convw = (const float*)d_in[4];
  const float* gamma = (const float*)d_in[5];
  const float* beta  = (const float*)d_in[6];
  const float* cm    = (const float*)d_in[7];
  const float* vleak = (const float*)d_in[8];
  const float* tau   = (const float*)d_in[9];
  const float* erev  = (const float*)d_in[10];
  const float* w1 = (const float*)d_in[11]; const float* b1 = (const float*)d_in[12];
  const float* w2 = (const float*)d_in[13]; const float* b2 = (const float*)d_in[14];
  const float* w3 = (const float*)d_in[15]; const float* b3 = (const float*)d_in[16];
  float* out = (float*)d_out;
  float* ws  = (float*)d_ws;

  dim3 gridC(IMG / TX, IMG / TY, 8);
  conv_ltc_kernel<<<gridC, 256, 0, stream>>>(x, convw, gamma, beta, cm, vleak, tau, erev, out);

  // preds ordered: scale0 = y3 (C=192, 64x64), scale1 = y2 (384, 32x32), scale2 = y1 (768, 16x16)
  head_kernel<<<dim3(32, 1, 8), 256, 0, stream>>>(y3, w3, b3, 192, 4096, ws);
  head_kernel<<<dim3(8, 1, 8),  256, 0, stream>>>(y2, w2, b2, 384, 1024, ws + 688128);
  head_kernel<<<dim3(2, 1, 8),  256, 0, stream>>>(y1, w1, b1, 768, 256,  ws + 860160);

  decode_kernel<<<(8 * 16128 + 255) / 256, 256, 0, stream>>>(ws, out + ENC_ELEMS);
}
